// MatMul_62706522522095
// MI455X (gfx1250) — matrix operand order repaired and checked by reading, not compiled, not run on hardware
//
#include <hip/hip_runtime.h>

// Problem constants (from the reference).
#define B_  32
#define L_  4096
#define D_  512
#define TM  128     // output tile rows per block
#define TN  128     // output tile cols per block
#define TK  32      // K chunk = one bf16 WMMA K-step
#define NK  (D_ / TK)
#define LDSW 40     // shorts per LDS row (32 data + 8 pad) = 80 bytes, 16B aligned

typedef __attribute__((ext_vector_type(16))) __bf16        v16bf;
typedef __attribute__((ext_vector_type(4)))  __bf16        v4bf;
typedef __attribute__((ext_vector_type(8)))  float         v8f;
typedef __attribute__((ext_vector_type(4)))  float         v4f;
typedef __attribute__((ext_vector_type(4)))  unsigned int  v4u;

union FragU { v16bf v; v4u q[2]; };

// Native f32 -> bf16 (RNE) via hardware cvt path; pack 4 into one 64-bit word.
__device__ __forceinline__ unsigned long long pack4(v4f v) {
    union { v4bf b; unsigned long long u; } cv;
    cv.b = __builtin_convertvector(v, v4bf);
    return cv.u;
}

__device__ __forceinline__ unsigned short bfbits(float f) {
    union { __bf16 h; unsigned short u; } cv;
    cv.h = (__bf16)f;
    return cv.u;
}

__global__ __launch_bounds__(256)
void ragged_bmm_wmma_bf16(const float* __restrict__ x,
                          const long long* __restrict__ lengths,
                          const float* __restrict__ bm,
                          float* __restrict__ out) {
    // Double-buffered bf16 staging tiles. A: [row m][k], Bt: [col n][k].
    __shared__ __align__(16) unsigned short As[2][TM][LDSW];
    __shared__ __align__(16) unsigned short Bs[2][TN][LDSW];

    const int t     = threadIdx.x;
    const int batch = blockIdx.z;
    const int gm0   = blockIdx.y * TM;
    const int gn0   = blockIdx.x * TN;
    const long long len = lengths[batch];

    // Global-load assignment (fixed per thread across K-steps).
    const int arow = t >> 3;         // 0..31, +q*32  -> A rows
    const int acol = (t & 7) * 4;    // 0..28         -> A cols (K), float4
    const int bk   = t >> 5;         // 0..7,  +q*8   -> B rows (K)
    const int bn   = (t & 31) * 4;   // 0..124        -> B cols (N), float4

    const float* xb = x  + (size_t)batch * L_ * D_;
    const float* bb = bm + (size_t)batch * D_ * D_;

    v4f aReg[4], bReg[4];

    auto issue_loads = [&](int kc) {
        const int k0 = kc * TK;
#pragma unroll
        for (int q = 0; q < 4; ++q) {
            const int r = q * 32 + arow;   // always in-bounds; mask values later
            aReg[q] = *(const v4f*)(xb + (size_t)(gm0 + r) * D_ + k0 + acol);
        }
#pragma unroll
        for (int q = 0; q < 4; ++q) {
            const int k = q * 8 + bk;
            bReg[q] = *(const v4f*)(bb + (size_t)(k0 + k) * D_ + gn0 + bn);
        }
    };

    auto stage = [&](int buf) {
#pragma unroll
        for (int q = 0; q < 4; ++q) {
            const int r = q * 32 + arow;
            const bool valid = (long long)(gm0 + r) < len;   // ragged row mask
            unsigned long long pv = pack4(aReg[q]);
            *(unsigned long long*)&As[buf][r][acol] = valid ? pv : 0ull;
        }
#pragma unroll
        for (int q = 0; q < 4; ++q) {
            const int k = q * 8 + bk;
            Bs[buf][bn + 0][k] = bfbits(bReg[q].x);   // transpose into Bt[n][k]
            Bs[buf][bn + 1][k] = bfbits(bReg[q].y);
            Bs[buf][bn + 2][k] = bfbits(bReg[q].z);
            Bs[buf][bn + 3][k] = bfbits(bReg[q].w);
        }
    };

    // Wave tiling: 4 (M) x 2 (N) waves; each wave owns 32x64 = 2x4 WMMA tiles.
    const int wave = t >> 5;
    const int lane = t & 31;
    const int wm = wave >> 1;
    const int wn = wave & 1;
    const int lr = lane & 15;        // row/col within a 16x16 tile
    const int kh = lane >> 4;        // K-half selector for A/B fragments

    v8f acc[2][4] = {};

    issue_loads(0);
    stage(0);
    __syncthreads();

    int buf = 0;
    for (int kc = 0; kc < NK; ++kc) {
        if (kc + 1 < NK) issue_loads(kc + 1);    // overlap global loads with WMMA

        FragU a[2], bf[4];
#pragma unroll
        for (int tm = 0; tm < 2; ++tm) {
            // A 16x32 bf16 frag: V0-3 -> K = kh*8..+7, V4-7 -> K = 16+kh*8..+7
            const unsigned short* rowp = &As[buf][wm * 32 + tm * 16 + lr][0];
            a[tm].q[0] = *(const v4u*)(rowp + kh * 8);
            a[tm].q[1] = *(const v4u*)(rowp + 16 + kh * 8);
        }
#pragma unroll
        for (int tn = 0; tn < 4; ++tn) {
            // B 32x16 bf16 frag: lane=n, V0-3 -> K = kh*8..+7, V4-7 -> K = 16+kh*8..+7
            const unsigned short* rowp = &Bs[buf][wn * 64 + tn * 16 + lr][0];
            bf[tn].q[0] = *(const v4u*)(rowp + kh * 8);
            bf[tn].q[1] = *(const v4u*)(rowp + 16 + kh * 8);
        }
#pragma unroll
        for (int tm = 0; tm < 2; ++tm)
#pragma unroll
            for (int tn = 0; tn < 4; ++tn)
                acc[tm][tn] = __builtin_amdgcn_wmma_f32_16x16x32_bf16(
                    false, a[tm].v, false, bf[tn].v,
                    (short)0, acc[tm][tn], false, false);

        if (kc + 1 < NK) {
            stage(buf ^ 1);          // write the other buffer (no readers now)
            __syncthreads();
            buf ^= 1;
        }
    }

    // Epilogue: C/D layout -> VGPR r: lanes 0-15 row r, lanes 16-31 row r+8.
    float* ob = out + (size_t)batch * L_ * D_;
#pragma unroll
    for (int tm = 0; tm < 2; ++tm) {
#pragma unroll
        for (int tn = 0; tn < 4; ++tn) {
            const int n     = gn0 + wn * 64 + tn * 16 + lr;
            const int mbase = gm0 + wm * 32 + tm * 16 + kh * 8;
#pragma unroll
            for (int r = 0; r < 8; ++r)
                ob[(size_t)(mbase + r) * D_ + n] = acc[tm][tn][r];
        }
    }
}

extern "C" void kernel_launch(void* const* d_in, const int* in_sizes, int n_in,
                              void* d_out, int out_size, void* d_ws, size_t ws_size,
                              hipStream_t stream) {
    const float*     x       = (const float*)d_in[0];
    const long long* lengths = (const long long*)d_in[1];
    const float*     bm      = (const float*)d_in[2];
    float*           out     = (float*)d_out;

    dim3 grid(D_ / TN, L_ / TM, B_);   // (4, 32, 32)
    dim3 block(256);
    hipLaunchKernelGGL(ragged_bmm_wmma_bf16, grid, block, 0, stream,
                       x, lengths, bm, out);
}
